// JModel_52080773431572
// MI455X (gfx1250) — matrix operand order repaired and checked by reading, not compiled, not run on hardware
//
#include <hip/hip_runtime.h>
#include <math.h>

// ---------------------------------------------------------------------------
// Model dims
// ---------------------------------------------------------------------------
#define N_ROWS 4096
#define SYN    64
#define SEM    128
#define FEAT   519          // 64 + 128 + 128 + 64 + 128 + 7
#define KX     544          // FEAT padded up to multiple of 32
#define HID    1024
#define OUTN   4096
#define BAG    8

typedef __attribute__((ext_vector_type(16))) __bf16 bf16x16;
typedef __attribute__((ext_vector_type(8)))  __bf16 bf16x8;
typedef __attribute__((ext_vector_type(8)))  float  f32x8;

union Frag16 { bf16x16 v; bf16x8 h[2]; };

// ---------------------------------------------------------------------------
// Kernel 1: assemble X [N_ROWS x KX] in bf16.
// Feature layout (matches jnp.concatenate order):
//   [0,64) cat_a | [64,192) hva | [192,320) hvf | [320,384) cat_l
//   [384,512) hvl | [512,519) d_onehot | [519,544) zero pad
// One block (128 threads) per row.
// ---------------------------------------------------------------------------
__device__ __forceinline__ float bag_sum(const int* __restrict__ ix,
                                         const float* __restrict__ emb,
                                         const float* __restrict__ top,
                                         int row, int t) {
    int idx[BAG];
#pragma unroll
    for (int j = 0; j < BAG; ++j) idx[j] = ix[row * BAG + j];
    float acc = top[(size_t)row * SEM + t];
#pragma unroll
    for (int j = 0; j < BAG; ++j) {
        bool dup = false;
#pragma unroll
        for (int k = 0; k < j; ++k) dup |= (idx[k] == idx[j]);
        if (!dup) acc += emb[(size_t)idx[j] * SEM + t];
    }
    return acc;
}

__global__ __launch_bounds__(128) void prep_x_kernel(
    const float* __restrict__ d_onehot,
    const int*   __restrict__ cat_a_ix,
    const int*   __restrict__ hva_ix,
    const int*   __restrict__ hvf_ix,
    const int*   __restrict__ cat_l_ix,
    const int*   __restrict__ hvl_ix,
    const float* __restrict__ hva_top,
    const float* __restrict__ hvf_top,
    const float* __restrict__ hvl_top,
    const float* __restrict__ cat_embeds,
    const float* __restrict__ hvec_embeds,
    __bf16* __restrict__ X)
{
    const int row = blockIdx.x;
    const int t   = threadIdx.x;           // 0..127
    __bf16* xr = X + (size_t)row * KX;

    if (t < SYN) {
        xr[t]       = (__bf16)cat_embeds[(size_t)cat_a_ix[row] * SYN + t];
        xr[320 + t] = (__bf16)cat_embeds[(size_t)cat_l_ix[row] * SYN + t];
    }
    xr[64  + t] = (__bf16)bag_sum(hva_ix, hvec_embeds, hva_top, row, t);
    xr[192 + t] = (__bf16)bag_sum(hvf_ix, hvec_embeds, hvf_top, row, t);
    xr[384 + t] = (__bf16)bag_sum(hvl_ix, hvec_embeds, hvl_top, row, t);
    if (t < 7)       xr[512 + t] = (__bf16)d_onehot[row * 7 + t];
    else if (t < 32) xr[512 + t] = (__bf16)0.0f;   // pad 519..543
}

// ---------------------------------------------------------------------------
// Weight conversion kernels (f32 -> bf16, W1 padded K=519 -> 544)
// ---------------------------------------------------------------------------
__global__ __launch_bounds__(256) void conv_w1_kernel(const float* __restrict__ W1,
                                                      __bf16* __restrict__ W1b)
{
    int i = blockIdx.x * 256 + threadIdx.x;          // over HID*KX
    if (i >= HID * KX) return;
    int r = i / KX, c = i % KX;
    W1b[i] = (__bf16)((c < FEAT) ? W1[(size_t)r * FEAT + c] : 0.0f);
}

__global__ __launch_bounds__(256) void conv_w2_kernel(const float* __restrict__ W2,
                                                      __bf16* __restrict__ W2b)
{
    int i = blockIdx.x * 256 + threadIdx.x;          // over OUTN*HID
    W2b[i] = (__bf16)W2[i];
}

// ---------------------------------------------------------------------------
// NT GEMM: C[M x Nout] = A[M x K] * B[Nout x K]^T + bias, bf16 operands,
// f32 WMMA accumulation. K multiple of 32.
//
// Block tile: 256(M) x 64(N); the 8 waves of a block stack along M and all
// share the SAME 64 B columns — every wave issues identical B addresses per
// K chunk, so B fragments broadcast out of the WGP cache and L2 traffic is
// dominated by the (unshared) A stream. Wave tile: 32(M) x 64(N) =
// 2 A frags x 4 B frags -> 8 v_wmma per 32-wide K step, register ping-pong
// double buffering keeps the next chunk's loads in flight under the WMMAs.
//
// A fragment (16-bit A 16x32 ISA layout): lane (lrow=l&15, half=l>>4)
//   holds row m+lrow; V0..3 = K kbase+half*8..+7, V4..7 = +16.
// B fragment (32x16 B layout): lane holds column n+lrow; V0..3 = K
//   kbase+half*8..+7, V4..7 = +16 (same K order as A).
// C/D: VGPR r, lane l -> (m + r + half*8, n + lrow).
// ---------------------------------------------------------------------------
template <bool RELU_BF16_OUT, int TILES_N>
__global__ __launch_bounds__(256) void gemm_wmma_kernel(
    const __bf16* __restrict__ A,
    const __bf16* __restrict__ B,
    const float*  __restrict__ bias,
    void*         __restrict__ Cout,
    int K, int Nout)
{
    const int wid  = threadIdx.x >> 5;
    const int lane = threadIdx.x & 31;
    const int m0   = (blockIdx.x / TILES_N) * 256 + wid * 32;  // waves stack in M
    const int n0   = (blockIdx.x % TILES_N) * 64;              // shared across waves
    const int half = lane >> 4;
    const int lrow = lane & 15;

    const __bf16* aptr0 = A + (size_t)(m0 + lrow) * K + half * 8;        // rows m0..m0+15
    const __bf16* aptr1 = aptr0 + (size_t)16 * K;                         // rows m0+16..m0+31
    const __bf16* bptr  = B + (size_t)(n0 + lrow) * K + half * 8;

    f32x8 acc0[4] = {f32x8{}, f32x8{}, f32x8{}, f32x8{}};
    f32x8 acc1[4] = {f32x8{}, f32x8{}, f32x8{}, f32x8{}};

    auto loadA = [&](Frag16* dst, int k0) {
        dst[0].h[0] = *(const bf16x8*)(aptr0 + k0);
        dst[0].h[1] = *(const bf16x8*)(aptr0 + k0 + 16);
        dst[1].h[0] = *(const bf16x8*)(aptr1 + k0);
        dst[1].h[1] = *(const bf16x8*)(aptr1 + k0 + 16);
    };
    auto loadB = [&](Frag16* dst, int k0) {
#pragma unroll
        for (int t = 0; t < 4; ++t) {
            const __bf16* bp = bptr + (size_t)t * 16 * K + k0;
            dst[t].h[0] = *(const bf16x8*)(bp);
            dst[t].h[1] = *(const bf16x8*)(bp + 16);
        }
    };
    auto compute = [&](Frag16* a, Frag16* b) {
#pragma unroll
        for (int t = 0; t < 4; ++t) {
            acc0[t] = __builtin_amdgcn_wmma_f32_16x16x32_bf16(
                false, a[0].v, false, b[t].v, (short)0, acc0[t], false, false);
            acc1[t] = __builtin_amdgcn_wmma_f32_16x16x32_bf16(
                false, a[1].v, false, b[t].v, (short)0, acc1[t], false, false);
        }
    };

    Frag16 aA[2], bA[4], aB[2], bB[4];       // ping-pong buffers
    loadA(aA, 0);
    loadB(bA, 0);

    int k0 = 0;
    for (; k0 + 64 <= K; k0 += 64) {
        loadA(aB, k0 + 32);                  // issue next-chunk loads first
        loadB(bB, k0 + 32);
        compute(aA, bA);                     // waits tolerate in-flight next chunk
        if (k0 + 64 < K) {
            loadA(aA, k0 + 64);
            loadB(bA, k0 + 64);
        }
        compute(aB, bB);
    }
    if (k0 < K) compute(aA, bA);             // odd chunk count (K/32 == 17 for GEMM1)

    // ---- epilogue ----
#pragma unroll
    for (int t = 0; t < 4; ++t) {
        const int   n  = n0 + t * 16 + lrow;
        const float bv = bias[n];
#pragma unroll
        for (int r = 0; r < 8; ++r) {
            const int m_a = m0 + r + half * 8;        // rows of acc0
            const int m_b = m_a + 16;                 // rows of acc1
            float va = acc0[t][r] + bv;
            float vb = acc1[t][r] + bv;
            if (RELU_BF16_OUT) {
                va = va > 0.0f ? va : 0.0f;
                vb = vb > 0.0f ? vb : 0.0f;
                ((__bf16*)Cout)[(size_t)m_a * Nout + n] = (__bf16)va;
                ((__bf16*)Cout)[(size_t)m_b * Nout + n] = (__bf16)vb;
            } else {
                ((float*)Cout)[(size_t)m_a * Nout + n] = va;
                ((float*)Cout)[(size_t)m_b * Nout + n] = vb;
            }
        }
    }
}

// ---------------------------------------------------------------------------
// In-place row-wise log_softmax over OUTN columns. One block per row.
// ---------------------------------------------------------------------------
__global__ __launch_bounds__(256) void log_softmax_kernel(float* __restrict__ out)
{
    __shared__ float sdata[256];
    const int row = blockIdx.x;
    const int t   = threadIdx.x;
    float* x = out + (size_t)row * OUTN;

    float m = -INFINITY;
    for (int c = t; c < OUTN; c += 256) m = fmaxf(m, x[c]);
    sdata[t] = m; __syncthreads();
    for (int s = 128; s > 0; s >>= 1) {
        if (t < s) sdata[t] = fmaxf(sdata[t], sdata[t + s]);
        __syncthreads();
    }
    m = sdata[0]; __syncthreads();

    float sum = 0.0f;
    for (int c = t; c < OUTN; c += 256) sum += expf(x[c] - m);
    sdata[t] = sum; __syncthreads();
    for (int s = 128; s > 0; s >>= 1) {
        if (t < s) sdata[t] += sdata[t + s];
        __syncthreads();
    }
    const float lse = m + logf(sdata[0]);
    __syncthreads();

    for (int c = t; c < OUTN; c += 256) x[c] -= lse;
}

// ---------------------------------------------------------------------------
// Launcher
// ---------------------------------------------------------------------------
extern "C" void kernel_launch(void* const* d_in, const int* in_sizes, int n_in,
                              void* d_out, int out_size, void* d_ws, size_t ws_size,
                              hipStream_t stream)
{
    const float* d_onehot    = (const float*)d_in[0];
    const int*   cat_a_ix    = (const int*)  d_in[1];
    const int*   hva_ix      = (const int*)  d_in[2];
    const int*   hvf_ix      = (const int*)  d_in[3];
    const int*   cat_l_ix    = (const int*)  d_in[4];
    const int*   hvl_ix      = (const int*)  d_in[5];
    const float* hva_top     = (const float*)d_in[6];
    const float* hvf_top     = (const float*)d_in[7];
    const float* hvl_top     = (const float*)d_in[8];
    // d_in[9] = use_gpu scalar, unused
    const float* cat_embeds  = (const float*)d_in[10];
    const float* hvec_embeds = (const float*)d_in[11];
    const float* W1          = (const float*)d_in[12];
    const float* b1          = (const float*)d_in[13];
    const float* W2          = (const float*)d_in[14];
    const float* b2          = (const float*)d_in[15];
    float* out = (float*)d_out;

    // Workspace carve-up (256B aligned), total ~22.3 MB
    size_t off = 0;
    auto carve = [&](size_t bytes) -> char* {
        char* p = (char*)d_ws + off;
        off = (off + bytes + 255) & ~(size_t)255;
        return p;
    };
    __bf16* Xb  = (__bf16*)carve((size_t)N_ROWS * KX  * 2);
    __bf16* W1b = (__bf16*)carve((size_t)HID    * KX  * 2);
    __bf16* W2b = (__bf16*)carve((size_t)OUTN   * HID * 2);
    __bf16* Hb  = (__bf16*)carve((size_t)N_ROWS * HID * 2);
    (void)ws_size; (void)in_sizes; (void)n_in; (void)out_size;

    // 1) Build bf16 input matrix
    prep_x_kernel<<<N_ROWS, 128, 0, stream>>>(
        d_onehot, cat_a_ix, hva_ix, hvf_ix, cat_l_ix, hvl_ix,
        hva_top, hvf_top, hvl_top, cat_embeds, hvec_embeds, Xb);

    // 2) Weights -> bf16
    conv_w1_kernel<<<(HID * KX + 255) / 256, 256, 0, stream>>>(W1, W1b);
    conv_w2_kernel<<<(OUTN * HID) / 256, 256, 0, stream>>>(W2, W2b);

    // 3) H = relu(X @ W1^T + b1), bf16 out.
    //    Block tile 256x64: (4096/256) x (1024/64) = 256 blocks
    gemm_wmma_kernel<true, HID / 64><<<(N_ROWS / 256) * (HID / 64), 256, 0, stream>>>(
        Xb, W1b, b1, (void*)Hb, KX, HID);

    // 4) logits = H @ W2^T + b2, f32 out.
    //    Block tile 256x64: (4096/256) x (4096/64) = 1024 blocks
    gemm_wmma_kernel<false, OUTN / 64><<<(N_ROWS / 256) * (OUTN / 64), 256, 0, stream>>>(
        Hb, W2b, b2, (void*)out, HID, OUTN);

    // 5) In-place log_softmax per row
    log_softmax_kernel<<<N_ROWS, 256, 0, stream>>>(out);
}
